// MultiHeadAttention_90769838834285
// MI455X (gfx1250) — matrix operand order repaired and checked by reading, not compiled, not run on hardware
//
#include <hip/hip_runtime.h>

// ---------------------------------------------------------------------------
// MultiHeadAttention for MI455X (gfx1250): bf16 WMMA flash attention.
//   B=8, T=1024, C=1024, H=16, D=64
// Pipeline:
//   0) prep: x -> bf16 (streaming), W -> bf16 transposed [w][h][d][c]
//   1) qkv_proj: double-buffered TDM (tensor_load_to_lds) tile staging,
//      12 WMMAs per 32-wide K chunk per wave, outputs Q/K/V bf16 [B*H,T,D]
//   2) flash_attn: per (b,h), 64 query rows/block, online softmax over
//      32-wide KV tiles; K tile staged via global_load_async_to_lds_b128.
// ---------------------------------------------------------------------------

typedef __attribute__((ext_vector_type(16))) __bf16 v16bf;
typedef __attribute__((ext_vector_type(8)))  __bf16 v8bf;
typedef __attribute__((ext_vector_type(8)))  float  v8f;
typedef __attribute__((ext_vector_type(4)))  unsigned int u32x4;
typedef __attribute__((ext_vector_type(8)))  int i32x8;
typedef __attribute__((ext_vector_type(4)))  int i32x4;

union Frag16 {
    v16bf v;
    v8bf  h[2];
};

__device__ __forceinline__ __bf16 to_bf16(float f) {
    return static_cast<__bf16>(f);   // native gfx1250 f32->bf16 convert
}

constexpr int Bc = 8, Tc = 1024, Cc = 1024, Hc = 16, Dc = 64;
constexpr int BH = Bc * Hc;                           // 128
constexpr size_t QKV_ELEMS = (size_t)BH * Tc * Dc;    // 8,388,608 per tensor
constexpr size_t X_ELEMS   = (size_t)Bc * Tc * Cc;    // 8,388,608
constexpr size_t WT_ELEMS  = (size_t)3 * Hc * Dc * Cc;// 3,145,728

// ---------------------------------------------------------------------------
// TDM: issue a 2D bf16 tile load (tile 32 x 64 rows, row stride 1024 elems)
// global -> LDS via Tensor Data Mover. D# per CDNA5 ISA ch.8. Tracked by
// TENSORcnt. All descriptor inputs are wave-uniform.
// (This toolchain exposes the 6-arg builtin: g0, g1, g2, g3, g4, cpol.)
// ---------------------------------------------------------------------------
__device__ __forceinline__ void tdm_load_tile(unsigned lds_addr, unsigned long long ga,
                                              unsigned tdim1) {
    u32x4 g0;
    g0[0] = 1u;                                        // count=1, user mode
    g0[1] = lds_addr;                                  // LDS byte address
    g0[2] = (unsigned)(ga & 0xFFFFFFFFull);            // global_addr[31:0]
    g0[3] = (unsigned)((ga >> 32) & 0x01FFFFFFull)     // global_addr[56:32]
          | (2u << 30);                                // type=2 ("image")
    i32x8 g1;
    g1[0] = (int)(1u << 16);                           // data_size=1 -> 2 bytes
    g1[1] = (int)(1024u << 16);                        // tensor_dim0 = 1024 (low16 @ bit48)
    g1[2] = (int)(tdim1 << 16);                        // dim0 hi16=0 | tensor_dim1 low16
    g1[3] = (int)(32u << 16);                          // dim1 hi16=0 | tile_dim0 = 32
    g1[4] = (int)64u;                                  // tile_dim1 = 64, tile_dim2 = 0
    g1[5] = (int)1024u;                                // tensor_dim0_stride = 1024
    g1[6] = 0;
    g1[7] = 0;
    i32x4 gz4 = {0, 0, 0, 0};                          // 2D: groups 2/3 unused
    i32x8 gz8 = {0, 0, 0, 0, 0, 0, 0, 0};
    __builtin_amdgcn_tensor_load_to_lds(g0, g1, gz4, gz4, gz8, 0);
}

// ---------------------------------------------------------------------------
// Prep 0a: x f32 -> bf16 streaming convert (8 elems/thread)
// ---------------------------------------------------------------------------
__global__ __launch_bounds__(256) void cvt_x(const float* __restrict__ src,
                                             __bf16* __restrict__ dst, int n) {
    const int i = (blockIdx.x * 256 + threadIdx.x) * 8;
    if (i + 8 <= n) {
        const float4* s = reinterpret_cast<const float4*>(src + i);
        const float4 a = s[0], b = s[1];
        v8bf o;
        o[0] = to_bf16(a.x); o[1] = to_bf16(a.y); o[2] = to_bf16(a.z); o[3] = to_bf16(a.w);
        o[4] = to_bf16(b.x); o[5] = to_bf16(b.y); o[6] = to_bf16(b.z); o[7] = to_bf16(b.w);
        *reinterpret_cast<v8bf*>(dst + i) = o;
    }
}

// ---------------------------------------------------------------------------
// Prep 0b: W[h][c][d] f32 -> WT[w][h][d][c] bf16 (transpose + convert)
// grid = (H, 3), block = 256 (4 threads per d-row, 256 c-values each)
// ---------------------------------------------------------------------------
__global__ __launch_bounds__(256) void cvt_wT(const float* __restrict__ Wq,
                                              const float* __restrict__ Wk,
                                              const float* __restrict__ Wv,
                                              __bf16* __restrict__ WT) {
    const int h = blockIdx.x;
    const int w = blockIdx.y;
    const float* W = (w == 0) ? Wq : (w == 1) ? Wk : Wv;
    const float* Wh = W + (size_t)h * Cc * Dc;
    const int d  = threadIdx.x >> 2;
    const int c0 = (threadIdx.x & 3) * 256;
    __bf16* out = WT + (((size_t)w * Hc + h) * Dc + d) * Cc;
    for (int k = 0; k < 256; k += 8) {
        v8bf o;
        #pragma unroll
        for (int q = 0; q < 8; ++q)
            o[q] = to_bf16(Wh[(size_t)(c0 + k + q) * Dc + d]);
        *reinterpret_cast<v8bf*>(out + c0 + k) = o;
    }
}

// ---------------------------------------------------------------------------
// Kernel 1: fused per-head Q/K/V projection GEMM with TDM-staged tiles.
// Block tile 64(M) x 64(N==one head) x 3 weights; double-buffered LDS.
// grid = (M/64, H) = (128, 16), block = 128 threads (4 waves).
// Wave 0 DMAs the A tile; waves 1..3 DMA the Bq/Bk/Bv tiles.
// ---------------------------------------------------------------------------
__global__ __launch_bounds__(128) void qkv_proj(const __bf16* __restrict__ Xb,
                                                const __bf16* __restrict__ WT,
                                                __bf16* __restrict__ Qo,
                                                __bf16* __restrict__ Ko,
                                                __bf16* __restrict__ Vo) {
    __shared__ __bf16 Alds[2][64 * 32];      // [buf][m][c]
    __shared__ __bf16 Bt[2][3][64 * 32];     // [buf][w][d][c]

    const int tid  = threadIdx.x;
    const int wv   = tid >> 5;
    const int lane = tid & 31;
    const int l16  = lane & 15;
    const int hi8  = (lane < 16) ? 0 : 8;   // A-frag K-base / C-row offset
    const int kbB  = (lane < 16) ? 0 : 16;  // unused; B-frags use hi8 too

    const int m0 = blockIdx.x * 64;         // global row in [0, B*T)
    const int h  = blockIdx.y;              // head == N tile (D=64)

    __bf16* const Os[3] = {Qo, Ko, Vo};

    // per-wave DMA source/dest (wave-uniform)
    unsigned long long gbase;
    unsigned ldsb[2];
    unsigned tdim1;
    if (wv == 0) {
        gbase   = (unsigned long long)(size_t)(Xb + (size_t)m0 * Cc);
        ldsb[0] = (unsigned)(size_t)&Alds[0][0];
        ldsb[1] = (unsigned)(size_t)&Alds[1][0];
        tdim1   = (unsigned)(Bc * Tc);
    } else {
        const int w = wv - 1;
        gbase   = (unsigned long long)(size_t)(WT + ((size_t)(w * Hc + h) * Dc) * Cc);
        ldsb[0] = (unsigned)(size_t)&Bt[0][w][0];
        ldsb[1] = (unsigned)(size_t)&Bt[1][w][0];
        tdim1   = (unsigned)Dc;
    }

    v8f acc[3][4] = {};

    // prologue DMA: chunk 0 -> buffer 0
    tdm_load_tile(ldsb[0], gbase, tdim1);

    for (int c0 = 0; c0 < Cc; c0 += 32) {
        const int buf = (c0 >> 5) & 1;
        // drain own DMA, then barrier publishes all 4 waves' tiles
        __builtin_amdgcn_s_wait_tensorcnt(0);
        __syncthreads();
        // overlap: kick DMA for next chunk into the other buffer
        if (c0 + 32 < Cc)
            tdm_load_tile(ldsb[buf ^ 1], gbase + (unsigned long long)(c0 + 32) * 2u, tdim1);

        // ---- one A fragment, reused for 12 WMMAs ----
        Frag16 af;
        {
            const int row = wv * 16 + l16;
            af.h[0] = *reinterpret_cast<const v8bf*>(&Alds[buf][row * 32 + hi8]);
            af.h[1] = *reinterpret_cast<const v8bf*>(&Alds[buf][row * 32 + hi8 + 16]);
        }
        #pragma unroll
        for (int w = 0; w < 3; ++w) {
            #pragma unroll
            for (int nt = 0; nt < 4; ++nt) {
                Frag16 bf;
                const int col = nt * 16 + l16;      // d index
                bf.h[0] = *reinterpret_cast<const v8bf*>(&Bt[buf][w][col * 32 + hi8]);
                bf.h[1] = *reinterpret_cast<const v8bf*>(&Bt[buf][w][col * 32 + hi8 + 16]);
                acc[w][nt] = __builtin_amdgcn_wmma_f32_16x16x32_bf16(
                    false, af.v, false, bf.v, (short)0, acc[w][nt], false, false);
            }
        }
    }

    // ---- epilogue: write bf16 Q/K/V in [B*H, T, D] ----
    #pragma unroll
    for (int w = 0; w < 3; ++w) {
        #pragma unroll
        for (int nt = 0; nt < 4; ++nt) {
            #pragma unroll
            for (int i = 0; i < 8; ++i) {
                const int m = m0 + wv * 16 + i + hi8;     // global row in B*T
                const int b = m >> 10;
                const int t = m & (Tc - 1);
                const int d = nt * 16 + l16;
                Os[w][((size_t)(b * Hc + h) * Tc + t) * Dc + d] =
                    to_bf16(acc[w][nt][i]);
            }
        }
    }
}

// ---------------------------------------------------------------------------
// Kernel 2: causal flash attention.
// grid = (T/64, B*H) = (16, 128), block = 128 threads (4 waves).
// Wave w owns query rows [q0 + 16w, q0 + 16w + 16).
// ---------------------------------------------------------------------------
__global__ __launch_bounds__(128) void flash_attn(const __bf16* __restrict__ Q,
                                                  const __bf16* __restrict__ K,
                                                  const __bf16* __restrict__ V,
                                                  float* __restrict__ O) {
    __shared__ __bf16 Klds[32 * 64];      // [kv][d]  (filled by async DMA)
    __shared__ __bf16 Vt[64 * 32];        // [d][kv]
    __shared__ __bf16 Plds[4][16 * 32];   // per-wave P tile [m][kv]

    const int tid  = threadIdx.x;
    const int wv   = tid >> 5;
    const int lane = tid & 31;
    const int l16  = lane & 15;
    const int hi8  = (lane < 16) ? 0 : 8;
    const int kbB  = (lane < 16) ? 0 : 16;

    const int bh = blockIdx.y;
    const int q0 = blockIdx.x * 64;
    const int qw = q0 + wv * 16;          // this wave's first query row

    const __bf16* Qp = Q + (size_t)bh * Tc * Dc;
    const __bf16* Kp = K + (size_t)bh * Tc * Dc;
    const __bf16* Vp = V + (size_t)bh * Tc * Dc;

    // Q fragments, loaded once: two 16x32 A-frags over d = [0,32) and [32,64)
    Frag16 qf[2];
    {
        const int t = qw + l16;
        #pragma unroll
        for (int c = 0; c < 2; ++c) {
            const int dk = c * 32;
            qf[c].h[0] = *reinterpret_cast<const v8bf*>(&Qp[(size_t)t * Dc + dk + hi8]);
            qf[c].h[1] = *reinterpret_cast<const v8bf*>(&Qp[(size_t)t * Dc + dk + hi8 + 16]);
        }
    }

    v8f   oacc[4] = {};
    float mst[8], lst[8];
    #pragma unroll
    for (int i = 0; i < 8; ++i) { mst[i] = -1e30f; lst[i] = 0.f; }

    const float sm_scale = 0.125f;        // D^-0.5
    const int nblk = q0 / 32 + 2;         // causal bound for this block

    for (int j = 0; j < nblk; ++j) {
        const int kv0 = j * 32;
        const int r   = tid >> 2;         // 0..31 (kv within tile)
        const int d0  = (tid & 3) * 16;   // 0,16,32,48

        // ---- K tile: async DMA, global -> LDS (32B per thread = 4KB tile) ----
        {
            const unsigned ldsK = (unsigned)(size_t)&Klds[r * 64 + d0];
            const unsigned goff = (unsigned)((unsigned)((kv0 + r) * Dc + d0) * 2u);
            asm volatile(
                "global_load_async_to_lds_b128 %0, %1, %2\n\t"
                "global_load_async_to_lds_b128 %0, %1, %2 offset:16"
                :: "v"(ldsK), "v"(goff), "s"(Kp)
                : "memory");
        }
        // ---- V tile: transpose into LDS (manual; async cannot transpose) ----
        {
            const __bf16* vs = &Vp[(size_t)(kv0 + r) * Dc + d0];
            #pragma unroll
            for (int q = 0; q < 16; ++q) Vt[(d0 + q) * 32 + r] = vs[q];
        }
        // prefetch next KV tile into cache (global_prefetch_b8); L2-resident set
        if (j + 1 < nblk) {
            __builtin_prefetch(&Kp[(size_t)(kv0 + 32 + r) * Dc + d0], 0, 0);
            __builtin_prefetch(&Vp[(size_t)(kv0 + 32 + r) * Dc + d0], 0, 0);
        }
        // drain this wave's async DMA, then publish via workgroup barrier
        asm volatile("s_wait_asynccnt 0" ::: "memory");
        __syncthreads();

        // ---- S = Q K^T : two 16x16 n-tiles, K-dim 64 = 2 WMMAs each ----
        v8f s[2] = {};
        #pragma unroll
        for (int nt = 0; nt < 2; ++nt) {
            #pragma unroll
            for (int c = 0; c < 2; ++c) {
                const int dk  = c * 32;
                const int col = nt * 16 + l16;   // key index within tile
                Frag16 kf;
                kf.h[0] = *reinterpret_cast<const v8bf*>(&Klds[col * 64 + dk + hi8]);
                kf.h[1] = *reinterpret_cast<const v8bf*>(&Klds[col * 64 + dk + hi8 + 16]);
                s[nt] = __builtin_amdgcn_wmma_f32_16x16x32_bf16(
                    false, qf[c].v, false, kf.v, (short)0, s[nt], false, false);
            }
        }

        // ---- scale, causal mask, online softmax (rows live in 16-lane halves)
        #pragma unroll
        for (int i = 0; i < 8; ++i) {
            const int row = qw + i + hi8;
            float a0 = s[0][i] * sm_scale;
            float a1 = s[1][i] * sm_scale;
            if (kv0 + l16      > row) a0 = -1e30f;
            if (kv0 + 16 + l16 > row) a1 = -1e30f;

            float mx = fmaxf(a0, a1);
            mx = fmaxf(mx, __shfl_xor(mx, 1, 32));
            mx = fmaxf(mx, __shfl_xor(mx, 2, 32));
            mx = fmaxf(mx, __shfl_xor(mx, 4, 32));
            mx = fmaxf(mx, __shfl_xor(mx, 8, 32));

            const float mn = fmaxf(mst[i], mx);
            const float sc = __expf(mst[i] - mn);
            mst[i] = mn;

            const float p0 = (a0 <= -1e29f) ? 0.f : __expf(a0 - mn);
            const float p1 = (a1 <= -1e29f) ? 0.f : __expf(a1 - mn);

            float ps = p0 + p1;
            ps += __shfl_xor(ps, 1, 32);
            ps += __shfl_xor(ps, 2, 32);
            ps += __shfl_xor(ps, 4, 32);
            ps += __shfl_xor(ps, 8, 32);
            lst[i] = lst[i] * sc + ps;

            #pragma unroll
            for (int dt = 0; dt < 4; ++dt) oacc[dt][i] *= sc;

            Plds[wv][(i + hi8) * 32 + l16]      = to_bf16(p0);
            Plds[wv][(i + hi8) * 32 + 16 + l16] = to_bf16(p1);
        }

        // wave-local LDS RAW: P stores (C layout) -> A-frag reads (cross-lane)
        asm volatile("s_wait_dscnt 0" ::: "memory");

        Frag16 pf;
        pf.h[0] = *reinterpret_cast<const v8bf*>(&Plds[wv][l16 * 32 + hi8]);
        pf.h[1] = *reinterpret_cast<const v8bf*>(&Plds[wv][l16 * 32 + hi8 + 16]);

        // ---- O += P V : 4 d-tiles of 16 ----
        #pragma unroll
        for (int dt = 0; dt < 4; ++dt) {
            Frag16 vf;
            const int col = dt * 16 + l16;       // d index
            vf.h[0] = *reinterpret_cast<const v8bf*>(&Vt[col * 32 + hi8]);
            vf.h[1] = *reinterpret_cast<const v8bf*>(&Vt[col * 32 + hi8 + 16]);
            oacc[dt] = __builtin_amdgcn_wmma_f32_16x16x32_bf16(
                false, pf.v, false, vf.v, (short)0, oacc[dt], false, false);
        }
        __syncthreads();
    }

    // ---- epilogue: out[b, t, h*64 + d] = O / l ----
    const int b = bh >> 4;
    const int h = bh & 15;
    #pragma unroll
    for (int i = 0; i < 8; ++i) {
        const float inv = (lst[i] > 0.f) ? (1.f / lst[i]) : 0.f;
        const int t = qw + i + hi8;
        #pragma unroll
        for (int dt = 0; dt < 4; ++dt) {
            const int d = dt * 16 + l16;
            O[((size_t)b * Tc + t) * (Hc * Dc) + h * Dc + d] = oacc[dt][i] * inv;
        }
    }
}

// ---------------------------------------------------------------------------
extern "C" void kernel_launch(void* const* d_in, const int* in_sizes, int n_in,
                              void* d_out, int out_size, void* d_ws, size_t ws_size,
                              hipStream_t stream) {
    (void)in_sizes; (void)n_in; (void)out_size; (void)ws_size;
    const float* x  = (const float*)d_in[0];
    const float* Wq = (const float*)d_in[1];
    const float* Wk = (const float*)d_in[2];
    const float* Wv = (const float*)d_in[3];
    float* out = (float*)d_out;

    __bf16* Qws = (__bf16*)d_ws;
    __bf16* Kws = Qws + QKV_ELEMS;
    __bf16* Vws = Kws + QKV_ELEMS;
    __bf16* Xb  = Vws + QKV_ELEMS;       // x in bf16
    __bf16* WT  = Xb + X_ELEMS;          // W transposed bf16 [w][h][d][c]

    dim3 blk256(256);
    cvt_x<<<dim3((unsigned)(X_ELEMS / 2048)), blk256, 0, stream>>>(x, Xb, (int)X_ELEMS);
    cvt_wT<<<dim3(Hc, 3), blk256, 0, stream>>>(Wq, Wk, Wv, WT);

    dim3 blk(128);
    dim3 pg((Bc * Tc) / 64, Hc);        // (128, 16)
    qkv_proj<<<pg, blk, 0, stream>>>(Xb, WT, Qws, Kws, Vws);

    dim3 ag(Tc / 64, BH);               // (16, 128)
    flash_attn<<<ag, blk, 0, stream>>>(Qws, Kws, Vws, out);
}
